// MoE_2594160247490
// MI455X (gfx1250) — matrix operand order repaired and checked by reading, not compiled, not run on hardware
//
#include <hip/hip_runtime.h>
#include <hip/hip_bf16.h>

// ---------------------------------------------------------------------------
// MoE top-2 FFN for MI455X (gfx1250, wave32, WMMA).
//   T=4096 tokens, H=1024, E=8 experts, I=4096, top-2.
// Round-3 design (gathered grouped-GEMM):
//   * Routing scatters tokens into per-expert compacted lists (idx/weight in
//     ws). FFN computes ONLY routed (token,expert) pairs: ~4x less WMMA work
//     and L2 weight traffic than dense-per-expert.
//   * Weights pre-converted to bf16 and TRANSPOSED once, so each WMMA
//     B-fragment is 16 contiguous bf16 = two global_load_b128, zero VALU.
//     bf16 weights (134MB) are L2-resident (192MB).
//   * Fused FFN per 16-token tile: 16x4096 bf16 intermediate in LDS (128KB of
//     the 320KB/WGP). 8 waves, 4 accumulators per K-step.
//   * Gathered x rows staged to LDS via per-lane async copies
//     (global_load_async_to_lds_b128 + s_wait_asynccnt).
// ---------------------------------------------------------------------------

typedef __attribute__((ext_vector_type(16))) __bf16 v16bf;
typedef __attribute__((ext_vector_type(8)))  float  v8f;

constexpr int BD = 2, S = 2048, Hd = 1024, E = 8, Id = 4096;
constexpr int T  = BD * S;          // 4096 tokens
constexpr int MT = 16;              // token tile (WMMA M)

__device__ __forceinline__ unsigned short f2bf(float f) {
    union { float f; unsigned u; } v; v.f = f;
    unsigned u = v.u;
    u += 0x7FFFu + ((u >> 16) & 1u);
    return (unsigned short)(u >> 16);
}

__global__ void zero_out(float* __restrict__ p, int n) {
    int i = blockIdx.x * 256 + threadIdx.x;
    if (i < n) p[i] = 0.0f;
}

// zero per-expert routing structures (counters, token lists, gate weights)
__global__ void zero_route(int* __restrict__ cnt, int* __restrict__ idx,
                           float* __restrict__ gw) {
    int i = blockIdx.x * 256 + threadIdx.x;   // E*T entries
    idx[i] = 0;
    gw[i]  = 0.0f;
    if (i < E) cnt[i] = 0;
}

// x[T][H] f32 -> xb[T][H] bf16
__global__ void cvt_x(const float* __restrict__ x, unsigned short* __restrict__ xb) {
    int i = (blockIdx.x * 256 + threadIdx.x) * 4;
    float4 v = *reinterpret_cast<const float4*>(x + i);
    ushort4 o;
    o.x = f2bf(v.x); o.y = f2bf(v.y); o.z = f2bf(v.z); o.w = f2bf(v.w);
    *reinterpret_cast<ushort4*>(xb + i) = o;
}

// Per-expert slice transpose + cvt: src [rows][cols] f32 -> dst [cols][rows] bf16.
__global__ void __launch_bounds__(256) transpose_cvt(const float* __restrict__ src,
                                                     unsigned short* __restrict__ dst,
                                                     int rows, int cols) {
    __shared__ float tile[32][33];
    const size_t slice = (size_t)rows * cols;
    const float* s = src + (size_t)blockIdx.y * slice;
    unsigned short* d = dst + (size_t)blockIdx.y * slice;
    const int tilesPerRow = cols >> 5;
    const int tr = blockIdx.x / tilesPerRow;
    const int tc = blockIdx.x % tilesPerRow;
    const int lr = threadIdx.x >> 5;
    const int lc = threadIdx.x & 31;
#pragma unroll
    for (int rr = lr; rr < 32; rr += 8)
        tile[rr][lc] = s[(size_t)(tr * 32 + rr) * cols + tc * 32 + lc];
    __syncthreads();
#pragma unroll
    for (int rr = lr; rr < 32; rr += 8)
        d[(size_t)(tc * 32 + rr) * rows + tr * 32 + lc] = f2bf(tile[lc][rr]);
}

// ---------------------------------------------------------------------------
// Routing: one wave32 per token; softmax + top-2 + renorm, then SCATTER the
// token into its two experts' compacted lists (atomic slot allocation).
// ---------------------------------------------------------------------------
__global__ void __launch_bounds__(256) moe_route(const float* __restrict__ x,
                                                 const float* __restrict__ Wg,
                                                 int* __restrict__ cnt,
                                                 int* __restrict__ idx,
                                                 float* __restrict__ gw) {
    const int wave = threadIdx.x >> 5;
    const int lane = threadIdx.x & 31;
    const int t    = blockIdx.x * 8 + wave;
    if (t >= T) return;

    float acc[E];
#pragma unroll
    for (int e = 0; e < E; ++e) acc[e] = 0.0f;
    const float* xr = x + (size_t)t * Hd;
    for (int h = lane; h < Hd; h += 32) {
        float xv = xr[h];
        const float4* wr = reinterpret_cast<const float4*>(Wg + (size_t)h * E);
        float4 w0 = wr[0], w1 = wr[1];
        acc[0] += xv * w0.x; acc[1] += xv * w0.y;
        acc[2] += xv * w0.z; acc[3] += xv * w0.w;
        acc[4] += xv * w1.x; acc[5] += xv * w1.y;
        acc[6] += xv * w1.z; acc[7] += xv * w1.w;
    }
#pragma unroll
    for (int e = 0; e < E; ++e) {
#pragma unroll
        for (int off = 16; off > 0; off >>= 1)
            acc[e] += __shfl_xor(acc[e], off, 32);
    }
    if (lane == 0) {
        float mx = acc[0];
#pragma unroll
        for (int e = 1; e < E; ++e) mx = fmaxf(mx, acc[e]);
        float p[E], s = 0.0f;
#pragma unroll
        for (int e = 0; e < E; ++e) { p[e] = __expf(acc[e] - mx); s += p[e]; }
        float inv = __builtin_amdgcn_rcpf(s);
#pragma unroll
        for (int e = 0; e < E; ++e) p[e] *= inv;
        int i0 = 0;
#pragma unroll
        for (int e = 1; e < E; ++e) if (p[e] > p[i0]) i0 = e;
        int i1 = (i0 == 0) ? 1 : 0;
#pragma unroll
        for (int e = 0; e < E; ++e) if (e != i0 && p[e] > p[i1]) i1 = e;
        float norm = __builtin_amdgcn_rcpf(p[i0] + p[i1]);

        int pos0 = atomicAdd(&cnt[i0], 1);
        idx[i0 * T + pos0] = t;
        gw [i0 * T + pos0] = p[i0] * norm;
        int pos1 = atomicAdd(&cnt[i1], 1);
        idx[i1 * T + pos1] = t;
        gw [i1 * T + pos1] = p[i1] * norm;
    }
}

// ---------------------------------------------------------------------------
// Gathered fused expert FFN. Grid = (T/16 tiles, E), 256 thr = 8 waves.
// Early-exits tiles beyond the expert's routed-token count (~4x fewer active).
// LDS (dynamic 160KB): xs[16][H] bf16 + hs[16][I] bf16.
// ---------------------------------------------------------------------------
__global__ void __launch_bounds__(256, 2) moe_ffn(const unsigned short* __restrict__ xb,
                                                  const unsigned short* __restrict__ W1t,
                                                  const float* __restrict__ b1,
                                                  const unsigned short* __restrict__ W2t,
                                                  const float* __restrict__ b2,
                                                  const int* __restrict__ cnt,
                                                  const int* __restrict__ idx,
                                                  const float* __restrict__ gw,
                                                  float* __restrict__ out) {
    const int e = blockIdx.y;
    if ((int)blockIdx.x * MT >= cnt[e]) return;   // uniform early-exit

    extern __shared__ unsigned short lds[];
    unsigned short* xs = lds;               // [MT][Hd] bf16 (32KB)
    unsigned short* hs = lds + MT * Hd;     // [MT][Id] bf16 (128KB)
    __shared__ int   tok[MT];
    __shared__ float wtok[MT];

    const int tid  = threadIdx.x;
    const int lane = tid & 31;
    const int wv   = tid >> 5;
    const int nlo  = lane & 15;
    const int khf  = lane >> 4;
    const int slot0 = e * T + blockIdx.x * MT;

    if (tid < MT) {                          // pad slots: tok=0, weight=0
        tok[tid]  = idx[slot0 + tid];
        wtok[tid] = gw[slot0 + tid];
    }
    __syncthreads();

    // ---- Async-GATHER 16 scattered token rows (2KB each) into LDS ----
    {
        const int row = tid >> 4, seg = tid & 15;        // 16 threads per row
        unsigned ldsOff = __builtin_amdgcn_groupstaticsize()
                        + (unsigned)(row * 2048 + seg * 16);
        unsigned long long ga =
            (unsigned long long)(uintptr_t)(xb + (size_t)tok[row] * Hd)
            + (unsigned)(seg * 16);
#pragma unroll
        for (int it = 0; it < 8; ++it) {     // 16 thr * 16B * 8 = 2KB per row
            asm volatile("global_load_async_to_lds_b128 %0, %1, off"
                         :: "v"(ldsOff + it * 256u), "v"(ga + it * 256u)
                         : "memory");
        }
        asm volatile("s_wait_asynccnt 0x0" ::: "memory");
    }
    __syncthreads();

    // ---------------- Phase 1: hdn[16, I] = silu(x @ W1[e] + b1) ----------
    const unsigned short* W1e = W1t + (size_t)e * Hd * Id;   // [Id][Hd]
    for (int g = 0; g < 8; ++g) {           // 8 groups * 4 N-tiles * 16 = 512/wave
        const int n0 = wv * 512 + g * 64;
        v8f acc0 = {}, acc1 = {}, acc2 = {}, acc3 = {};
        for (int k0 = 0; k0 < Hd; k0 += 32) {
            union { v16bf v; uint4 q[2]; } A;
            const unsigned short* ar = xs + nlo * Hd + k0 + khf * 16;
            A.q[0] = *reinterpret_cast<const uint4*>(ar);
            A.q[1] = *reinterpret_cast<const uint4*>(ar + 8);

            union { v16bf v; uint4 q[2]; } B0, B1, B2, B3;
            const uint4* bp0 = reinterpret_cast<const uint4*>(
                W1e + (size_t)(n0 + 0 * 16 + nlo) * Hd + k0 + khf * 16);
            const uint4* bp1 = reinterpret_cast<const uint4*>(
                W1e + (size_t)(n0 + 1 * 16 + nlo) * Hd + k0 + khf * 16);
            const uint4* bp2 = reinterpret_cast<const uint4*>(
                W1e + (size_t)(n0 + 2 * 16 + nlo) * Hd + k0 + khf * 16);
            const uint4* bp3 = reinterpret_cast<const uint4*>(
                W1e + (size_t)(n0 + 3 * 16 + nlo) * Hd + k0 + khf * 16);
            B0.q[0] = bp0[0]; B0.q[1] = bp0[1];
            B1.q[0] = bp1[0]; B1.q[1] = bp1[1];
            B2.q[0] = bp2[0]; B2.q[1] = bp2[1];
            B3.q[0] = bp3[0]; B3.q[1] = bp3[1];

            acc0 = __builtin_amdgcn_wmma_f32_16x16x32_bf16(false, A.v, false, B0.v, (short)0, acc0, false, false);
            acc1 = __builtin_amdgcn_wmma_f32_16x16x32_bf16(false, A.v, false, B1.v, (short)0, acc1, false, false);
            acc2 = __builtin_amdgcn_wmma_f32_16x16x32_bf16(false, A.v, false, B2.v, (short)0, acc2, false, false);
            acc3 = __builtin_amdgcn_wmma_f32_16x16x32_bf16(false, A.v, false, B3.v, (short)0, acc3, false, false);
        }
        v8f* accs[4] = { &acc0, &acc1, &acc2, &acc3 };
#pragma unroll
        for (int q = 0; q < 4; ++q) {
            const int ncol = n0 + q * 16 + nlo;
            const float b1v = b1[(size_t)e * Id + ncol];
            v8f a = *accs[q];
#pragma unroll
            for (int r = 0; r < 8; ++r) {
                float v  = a[r] + b1v;
                float sv = v * __builtin_amdgcn_rcpf(1.0f + __expf(-v));  // silu
                hs[(r + 8 * khf) * Id + ncol] = f2bf(sv);
            }
        }
    }
    __syncthreads();

    // ---------------- Phase 2: y[16, H] = hdn @ W2[e] + b2; scatter -------
    const unsigned short* W2e = W2t + (size_t)e * Id * Hd;   // [Hd][Id]
    for (int g = 0; g < 2; ++g) {           // 2 groups * 4 N-tiles * 16 = 128/wave
        const int h0 = wv * 128 + g * 64;
        v8f acc0 = {}, acc1 = {}, acc2 = {}, acc3 = {};
        for (int k0 = 0; k0 < Id; k0 += 32) {
            union { v16bf v; uint4 q[2]; } A;
            const unsigned short* ar = hs + nlo * Id + k0 + khf * 16;
            A.q[0] = *reinterpret_cast<const uint4*>(ar);
            A.q[1] = *reinterpret_cast<const uint4*>(ar + 8);

            union { v16bf v; uint4 q[2]; } B0, B1, B2, B3;
            const uint4* bp0 = reinterpret_cast<const uint4*>(
                W2e + (size_t)(h0 + 0 * 16 + nlo) * Id + k0 + khf * 16);
            const uint4* bp1 = reinterpret_cast<const uint4*>(
                W2e + (size_t)(h0 + 1 * 16 + nlo) * Id + k0 + khf * 16);
            const uint4* bp2 = reinterpret_cast<const uint4*>(
                W2e + (size_t)(h0 + 2 * 16 + nlo) * Id + k0 + khf * 16);
            const uint4* bp3 = reinterpret_cast<const uint4*>(
                W2e + (size_t)(h0 + 3 * 16 + nlo) * Id + k0 + khf * 16);
            B0.q[0] = bp0[0]; B0.q[1] = bp0[1];
            B1.q[0] = bp1[0]; B1.q[1] = bp1[1];
            B2.q[0] = bp2[0]; B2.q[1] = bp2[1];
            B3.q[0] = bp3[0]; B3.q[1] = bp3[1];

            acc0 = __builtin_amdgcn_wmma_f32_16x16x32_bf16(false, A.v, false, B0.v, (short)0, acc0, false, false);
            acc1 = __builtin_amdgcn_wmma_f32_16x16x32_bf16(false, A.v, false, B1.v, (short)0, acc1, false, false);
            acc2 = __builtin_amdgcn_wmma_f32_16x16x32_bf16(false, A.v, false, B2.v, (short)0, acc2, false, false);
            acc3 = __builtin_amdgcn_wmma_f32_16x16x32_bf16(false, A.v, false, B3.v, (short)0, acc3, false, false);
        }
        v8f* accs[4] = { &acc0, &acc1, &acc2, &acc3 };
#pragma unroll
        for (int q = 0; q < 4; ++q) {
            const int hcol = h0 + q * 16 + nlo;
            const float b2v = b2[(size_t)e * Hd + hcol];
            v8f a = *accs[q];
#pragma unroll
            for (int r = 0; r < 8; ++r) {
                const int m = r + 8 * khf;
                atomicAdd(&out[(size_t)tok[m] * Hd + hcol], (a[r] + b2v) * wtok[m]);
            }
        }
    }
}

extern "C" void kernel_launch(void* const* d_in, const int* in_sizes, int n_in,
                              void* d_out, int out_size, void* d_ws, size_t ws_size,
                              hipStream_t stream) {
    const float* x  = (const float*)d_in[0];
    const float* Wg = (const float*)d_in[1];
    const float* W1 = (const float*)d_in[2];
    const float* b1 = (const float*)d_in[3];
    const float* W2 = (const float*)d_in[4];
    const float* b2 = (const float*)d_in[5];
    float*       out = (float*)d_out;

    // workspace layout
    char* ws = (char*)d_ws;
    int*            cnt = (int*)ws;                                   // 64B slot
    int*            idx = (int*)(ws + 64);                            // E*T*4
    float*          gw  = (float*)(ws + 64 + (size_t)E * T * 4);      // E*T*4
    unsigned short* xb  = (unsigned short*)(ws + 64 + (size_t)E * T * 8);        // 8MB
    unsigned short* W1t = (unsigned short*)((char*)xb + (size_t)T * Hd * 2);     // 67MB
    unsigned short* W2t = (unsigned short*)((char*)W1t + (size_t)E * Hd * Id * 2);

    // 1) zero output + routing structures
    zero_out<<<(out_size + 255) / 256, 256, 0, stream>>>(out, out_size);
    zero_route<<<(E * T) / 256, 256, 0, stream>>>(cnt, idx, gw);

    // 2) one-time transforms: x->bf16 ; W1,W2 -> bf16 transposed (K-contig)
    cvt_x<<<(T * Hd / 4 + 255) / 256, 256, 0, stream>>>(x, xb);
    {   // W1 [E][Hd][Id] -> W1t [E][Id][Hd]
        dim3 g((Hd / 32) * (Id / 32), E);
        transpose_cvt<<<g, 256, 0, stream>>>(W1, W1t, Hd, Id);
    }
    {   // W2 [E][Id][Hd] -> W2t [E][Hd][Id]
        dim3 g((Id / 32) * (Hd / 32), E);
        transpose_cvt<<<g, 256, 0, stream>>>(W2, W2t, Id, Hd);
    }

    // 3) routing with per-expert compaction
    moe_route<<<T / 8, 256, 0, stream>>>(x, Wg, cnt, idx, gw);

    // 4) gathered grouped-GEMM FFN, 160KB dynamic LDS (CDNA5: 320KB/WGP)
    const size_t ldsBytes = (size_t)(MT * Hd + MT * Id) * sizeof(unsigned short);
    hipFuncSetAttribute((const void*)moe_ffn,
                        hipFuncAttributeMaxDynamicSharedMemorySize,
                        (int)ldsBytes);
    dim3 grid(T / MT, E);
    moe_ffn<<<grid, 256, ldsBytes, stream>>>(xb, W1t, b1, W2t, b2,
                                             cnt, idx, gw, out);
}
